// SupKLDiergence_47682726920268
// MI455X (gfx1250) — compile-verified, matrix operand order checked by reading, not run on hardware
//
#include <hip/hip_runtime.h>
#include <hip/hip_bf16.h>
#include <stdint.h>

// ---------- types for WMMA / TDM ----------
typedef __attribute__((ext_vector_type(16))) __bf16    v16bf;
typedef __attribute__((ext_vector_type(8)))  float     v8f;
typedef __attribute__((ext_vector_type(4)))  uint32_t  u32x4;
typedef __attribute__((ext_vector_type(4)))  int       i32x4;
typedef __attribute__((ext_vector_type(8)))  int       i32x8;

struct Pack32 { uint4 lo; uint4 hi; };          // 32 bytes == v16bf
static_assert(sizeof(Pack32) == sizeof(v16bf), "frag size");

// float -> bf16 (round to nearest even), stored as raw u16
__device__ __forceinline__ uint16_t f32_to_bf16(float f) {
    union { float f; uint32_t u; } c; c.f = f;
    uint32_t u = c.u;
    u += 0x7FFFu + ((u >> 16) & 1u);
    return (uint16_t)(u >> 16);
}

__device__ __forceinline__ float waveRedSum(float v) {
    #pragma unroll
    for (int off = 16; off > 0; off >>= 1) v += __shfl_xor(v, off, 32);
    return v;
}
__device__ __forceinline__ float waveRedMax(float v) {
    #pragma unroll
    for (int off = 16; off > 0; off >>= 1) v = fmaxf(v, __shfl_xor(v, off, 32));
    return v;
}

// Load one 16x32 bf16 fragment (ISA 16-bit A/B striping; base already includes
// the per-lane row offset and the lane-high K offset). Works for LDS pointers
// (addrspace-inferred -> ds_load_b128) and global pointers alike.
__device__ __forceinline__ v16bf load_frag(const uint16_t* base) {
    Pack32 p;
    p.lo = *(const uint4*)(base);        // K = khi .. khi+7
    p.hi = *(const uint4*)(base + 16);   // K = 16+khi .. 16+khi+7
    return __builtin_bit_cast(v16bf, p);
}

// ---------------- TDM: issue a 2-D bf16 tile load into LDS ----------------
// D# per CDNA5 ISA ch.8: group0 = {count, lds_addr, global_addr, type=2},
// group1 = {data_size=2B, tensor_dim0/1, tile_dim0/1, tensor_dim0_stride},
// groups 2/3 zero (2-D tile). Tracked with TENSORcnt.
__device__ __forceinline__ void tdm_load_2d(uint32_t lds_off,
                                            const uint16_t* gptr,
                                            int tensor_d0, int tensor_d1,
                                            int stride_elems,
                                            int tile_d0, int tile_d1) {
    const uint64_t ga = (uint64_t)(uintptr_t)gptr;
    u32x4 g0;
    g0[0] = 1u;                                            // count=1 (valid)
    g0[1] = lds_off;                                       // LDS byte address
    g0[2] = (uint32_t)ga;                                  // global addr lo
    g0[3] = (uint32_t)((ga >> 32) & 0x01FFFFFFu) | (2u << 30); // hi + type=2
    i32x8 g1;
    g1[0] = 1 << 16;                                       // data_size = 2B
    g1[1] = (tensor_d0 & 0xFFFF) << 16;                    // tensor_dim0 lo16
    g1[2] = ((tensor_d0 >> 16) & 0xFFFF) | ((tensor_d1 & 0xFFFF) << 16);
    g1[3] = ((tensor_d1 >> 16) & 0xFFFF) | ((tile_d0 & 0xFFFF) << 16);
    g1[4] = (tile_d1 & 0xFFFF);                            // tile_dim1, dim2=0
    g1[5] = stride_elems;                                  // dim0 stride lo32
    g1[6] = 0;                                             // stride hi, dim1 stride
    g1[7] = 0;
    const i32x4 z4 = {0, 0, 0, 0};
#if defined(__clang_major__) && (__clang_major__ >= 23)
    const i32x8 z8 = {0, 0, 0, 0, 0, 0, 0, 0};
    __builtin_amdgcn_tensor_load_to_lds(g0, g1, z4, z4, z8, 0);
#else
    __builtin_amdgcn_tensor_load_to_lds(g0, g1, z4, z4, 0);
#endif
}

// ---------------- kernel 0: zero the two f64 accumulators ----------------
__global__ void k_init(double* acc) {
    acc[0] = 0.0; acc[1] = 0.0;
}

// ---------------- kernel 1: row softmax -> bf16 P, bf16 LogP, f32 diag ----
__global__ void k_softmax(const float* __restrict__ feat,
                          uint16_t* __restrict__ P,
                          uint16_t* __restrict__ LP,
                          float* __restrict__ diag,
                          int d) {
    const int row = blockIdx.x;
    const float* x = feat + (size_t)row * (size_t)(2 * d);
    const int tid  = threadIdx.x;
    const int wid  = tid >> 5;
    __shared__ float red[8];

    float mx = -3.4e38f;
    for (int i = tid; i < d; i += 256) mx = fmaxf(mx, x[i]);
    mx = waveRedMax(mx);
    if ((tid & 31) == 0) red[wid] = mx;
    __syncthreads();
    float bmax = red[0];
    #pragma unroll
    for (int w = 1; w < 8; ++w) bmax = fmaxf(bmax, red[w]);
    __syncthreads();

    float s = 0.f;
    for (int i = tid; i < d; i += 256) s += __expf(x[i] - bmax);
    s = waveRedSum(s);
    if ((tid & 31) == 0) red[wid] = s;
    __syncthreads();
    float bsum = 0.f;
    #pragma unroll
    for (int w = 0; w < 8; ++w) bsum += red[w];
    __syncthreads();
    const float lse = __logf(bsum);

    uint16_t* prow = P  + (size_t)row * (size_t)d;
    uint16_t* lrow = LP + (size_t)row * (size_t)d;
    float dp = 0.f;
    for (int i = tid; i < d; i += 256) {
        float lp = x[i] - bmax - lse;
        float pp = __expf(lp);
        prow[i] = f32_to_bf16(pp);
        lrow[i] = f32_to_bf16(lp);
        dp += pp * lp;
    }
    dp = waveRedSum(dp);
    if ((tid & 31) == 0) red[wid] = dp;
    __syncthreads();
    if (tid == 0) {
        float t = 0.f;
        #pragma unroll
        for (int w = 0; w < 8; ++w) t += red[w];
        diag[row] = t;
    }
}

// Reduce one 16x16 f32 C/D tile with the same/diff label mask.
__device__ __forceinline__ void tile_reduce(const v8f& c, int i0, int j,
                                            const int* __restrict__ labels,
                                            int lj, float& sSame, float& sDiff) {
    #pragma unroll
    for (int r = 0; r < 8; ++r) {
        const int i = i0 + r;
        if (i != j) {
            const float v = c[r];
            if (labels[i] == lj) sSame += v; else sDiff += v;
        }
    }
}

// ---------------- kernel 2: TDM -> LDS double-buffered WMMA GEMM ----------
// Block tile: 64 (P rows) x 128 (LogP rows), 8 waves as 2x4 grid of 32x32
// macro-tiles (2x2 WMMA frags each). K advances in 64-wide slabs; wave 0
// issues TDM tensor loads for A (64x64) and B (128x64) into a double-buffered
// LDS stage, retires the older stage with s_wait_tensorcnt, and barriers
// publish it to all waves. Fragments are read from LDS (ds_load_b128).
#define SLAB_K     64
#define A_ELEMS    (64 * SLAB_K)            // 4096
#define B_ELEMS    (128 * SLAB_K)           // 8192
#define BUF_ELEMS  (A_ELEMS + B_ELEMS)      // 12288 (24 KB)

__global__ void __launch_bounds__(256)
k_gemm_reduce(const uint16_t* __restrict__ P,
              const uint16_t* __restrict__ LP,
              const int* __restrict__ labels,
              double* __restrict__ acc,
              int N, int d) {
    __shared__ __align__(16) uint16_t smem[2 * BUF_ELEMS];   // 48 KB

    const int lane = threadIdx.x & 31;
    const int wid  = threadIdx.x >> 5;
    const int nCB  = N >> 7;                     // 128-wide col blocks
    const int bR   = blockIdx.x / nCB;           // 64-row block of P
    const int bC   = blockIdx.x - bR * nCB;      // 128-row block of LogP
    const int wr   = wid >> 2;                   // wave row   {0,1}
    const int wc   = wid & 3;                    // wave col   {0..3}

    const int lrow = lane & 15;
    const int khi  = (lane >= 16) ? 8 : 0;       // 16-bit A/B K striping

    // LDS byte offsets of the two stage buffers (low 32 bits of the flat
    // shared-aperture address == LDS offset).
    const uint32_t ldsBase = (uint32_t)(uintptr_t)(&smem[0]);

    const uint16_t* gA = P  + (size_t)(bR * 64)  * (size_t)d;   // A tile rows
    const uint16_t* gB = LP + (size_t)(bC * 128) * (size_t)d;   // B tile rows

    const int nSlab = d / SLAB_K;

    // Prologue: stage slab 0 into buffer 0.
    if (wid == 0) {
        tdm_load_2d(ldsBase,                 gA, d, N, d, SLAB_K, 64);
        tdm_load_2d(ldsBase + A_ELEMS * 2,   gB, d, N, d, SLAB_K, 128);
    }

    // Per-wave LDS fragment bases (row pitch = SLAB_K elements = 128 B).
    const uint16_t* aRow0 = smem + (size_t)(wr * 32 + lrow) * SLAB_K + khi;
    const uint16_t* bRow0 = smem + A_ELEMS + (size_t)(wc * 32 + lrow) * SLAB_K + khi;

    v8f c00 = {}, c01 = {}, c10 = {}, c11 = {};
    for (int ks = 0; ks < nSlab; ++ks) {
        const int buf = ks & 1;
        if (wid == 0) {
            if (ks + 1 < nSlab) {
                const uint32_t off = ldsBase + (uint32_t)((buf ^ 1) * BUF_ELEMS * 2);
                const int k0 = (ks + 1) * SLAB_K;
                tdm_load_2d(off,                gA + k0, d, N, d, SLAB_K, 64);
                tdm_load_2d(off + A_ELEMS * 2,  gB + k0, d, N, d, SLAB_K, 128);
                __builtin_amdgcn_s_wait_tensorcnt(2);  // older stage done
            } else {
                __builtin_amdgcn_s_wait_tensorcnt(0);
            }
        }
        __syncthreads();                                   // stage published

        const uint16_t* aB = aRow0 + (size_t)buf * BUF_ELEMS;
        const uint16_t* bB = bRow0 + (size_t)buf * BUF_ELEMS;
        #pragma unroll
        for (int kk = 0; kk < 2; ++kk) {                   // 2 x K=32 steps
            const int ko = kk * 32;
            v16bf fa0 = load_frag(aB + ko);
            v16bf fa1 = load_frag(aB + ko + 16 * SLAB_K);
            v16bf fb0 = load_frag(bB + ko);
            v16bf fb1 = load_frag(bB + ko + 16 * SLAB_K);
            c00 = __builtin_amdgcn_wmma_f32_16x16x32_bf16(false, fa0, false, fb0,
                                                          (short)0, c00, false, false);
            c01 = __builtin_amdgcn_wmma_f32_16x16x32_bf16(false, fa0, false, fb1,
                                                          (short)0, c01, false, false);
            c10 = __builtin_amdgcn_wmma_f32_16x16x32_bf16(false, fa1, false, fb0,
                                                          (short)0, c10, false, false);
            c11 = __builtin_amdgcn_wmma_f32_16x16x32_bf16(false, fa1, false, fb1,
                                                          (short)0, c11, false, false);
        }
        __syncthreads();                                   // safe to overwrite
    }

    // Epilogue: masked pair reduction of the 32x32 macro-tile.
    const int jBase = bC * 128 + wc * 32 + (lane & 15);
    const int iBase = bR * 64  + wr * 32 + ((lane >= 16) ? 8 : 0);
    const int lj0 = labels[jBase];
    const int lj1 = labels[jBase + 16];
    float sSame = 0.f, sDiff = 0.f;
    tile_reduce(c00, iBase,      jBase,      labels, lj0, sSame, sDiff);
    tile_reduce(c01, iBase,      jBase + 16, labels, lj1, sSame, sDiff);
    tile_reduce(c10, iBase + 16, jBase,      labels, lj0, sSame, sDiff);
    tile_reduce(c11, iBase + 16, jBase + 16, labels, lj1, sSame, sDiff);

    sSame = waveRedSum(sSame);
    sDiff = waveRedSum(sDiff);

    __shared__ float sS[8], sD[8];
    if (lane == 0) { sS[wid] = sSame; sD[wid] = sDiff; }
    __syncthreads();
    if (threadIdx.x == 0) {
        double tS = 0.0, tD = 0.0;
        #pragma unroll
        for (int w = 0; w < 8; ++w) { tS += (double)sS[w]; tD += (double)sD[w]; }
        atomicAdd(&acc[0], tS);
        atomicAdd(&acc[1], tD);
    }
}

// ---------------- kernel 3: finalize -------------------------------------
__global__ void k_finalize(const float* __restrict__ diag,
                           const int* __restrict__ labels,
                           const double* __restrict__ acc,
                           float* __restrict__ out,
                           int N, int d) {
    int cnt[8];
    #pragma unroll
    for (int c = 0; c < 8; ++c) cnt[c] = 0;
    for (int i = 0; i < N; ++i) cnt[labels[i] & 7]++;
    double dS = 0.0, dD = 0.0;
    for (int i = 0; i < N; ++i) {
        const double di = (double)diag[i];
        const int c = cnt[labels[i] & 7];
        dS += di * (double)(c - 1);
        dD += di * (double)(N - c);
    }
    const double inv2d = 1.0 / (2.0 * (double)d);
    const double sameSum = (dS - acc[0]) * inv2d;
    const double diffSum = (dD - acc[1]) * inv2d;
    out[0] = (float)(sameSum / diffSum);
}

extern "C" void kernel_launch(void* const* d_in, const int* in_sizes, int n_in,
                              void* d_out, int out_size, void* d_ws, size_t ws_size,
                              hipStream_t stream) {
    const float* feat   = (const float*)d_in[0];   // (N, 2, d) f32
    const int*   labels = (const int*)d_in[1];     // (N,) i32
    const int N = in_sizes[1];                     // 2048
    const int d = in_sizes[0] / (2 * N);           // 4096

    uint16_t* P    = (uint16_t*)d_ws;                       // N*d bf16
    uint16_t* LP   = P + (size_t)N * (size_t)d;             // N*d bf16
    float*    diag = (float*)(LP + (size_t)N * (size_t)d);  // N f32
    double*   acc  = (double*)(diag + N);                   // 2 f64

    k_init<<<1, 1, 0, stream>>>(acc);
    k_softmax<<<N, 256, 0, stream>>>(feat, P, LP, diag, d);

    const int nBlocks = (N / 64) * (N / 128);      // 512
    k_gemm_reduce<<<nBlocks, 256, 0, stream>>>(P, LP, labels, acc, N, d);

    k_finalize<<<1, 1, 0, stream>>>(diag, labels, acc, (float*)d_out, N, d);
}
